// GraphTransformer_75935021794038
// MI455X (gfx1250) — matrix operand order repaired and checked by reading, not compiled, not run on hardware
//
#include <hip/hip_runtime.h>
#include <stdint.h>

#define NN 50000
#define EE 800000
#define DDIM 128
#define HH 8
#define CC 16
#define LL 3
#define GG 64
#define MT (NN / 16)    // 3125 row tiles (exact)
#define KTILES (DDIM / 32)  // 4
#define NTILES (DDIM / 16)  // 8

typedef __attribute__((ext_vector_type(16))) __bf16   v16bf;
typedef __attribute__((ext_vector_type(8)))  float    v8f;
typedef __attribute__((ext_vector_type(4)))  unsigned u32x4;
typedef __attribute__((ext_vector_type(4)))  float    f32x4;

// ---- helpers ---------------------------------------------------------------
__device__ __forceinline__ unsigned short f2bf(float f) {
    unsigned u = __float_as_uint(f);
    u += 0x7FFFu + ((u >> 16) & 1u);      // round-to-nearest-even
    return (unsigned short)(u >> 16);
}
// monotone float<->uint encoding for atomicMax-based segment max
__device__ __forceinline__ unsigned fenc(float f) {
    unsigned u = __float_as_uint(f);
    return (u & 0x80000000u) ? ~u : (u | 0x80000000u);
}
__device__ __forceinline__ float fdec(unsigned e) {
    unsigned u = (e & 0x80000000u) ? (e & 0x7FFFFFFFu) : ~e;
    return __uint_as_float(u);
}

// ---- pack activations (f32 row-major -> WMMA A-layout bf16) ----------------
// A 16x32 bf16 layout (ISA 7.12.2): lane = (half<<4)|m, VGPR j holds K pair:
//   K = (j>=4)*16 + half*8 + (j&3)*2
// packed[(mt*4+kt)*256 + lane*8 + j] = dword{bf(x[m,K]), bf(x[m,K+1])}
__global__ void k_pack_a(const float* __restrict__ x, unsigned* __restrict__ ap) {
    int tid  = blockIdx.x * 256 + threadIdx.x;   // NN*DDIM/2 = 3,200,000 exact
    int blk  = tid >> 8;
    int w    = tid & 255;
    int lane = w >> 3, j = w & 7;
    int mt = blk >> 2, kt = blk & 3;
    int row = mt * 16 + (lane & 15);
    int col = kt * 32 + ((j >> 2) << 4) + ((lane >> 4) << 3) + ((j & 3) << 1);
    float a0 = x[row * DDIM + col];
    float a1 = x[row * DDIM + col + 1];
    ap[tid] = (unsigned)f2bf(a0) | ((unsigned)f2bf(a1) << 16);
}

// ---- pack weights (4 matrices -> WMMA B-layout bf16) -----------------------
// B 32x16 bf16: lane = (khalf<<4)|n, VGPR j holds K = (j>=4)*16 + khalf*8 + (j&3)*2, +1
__global__ void k_pack_w(const float* __restrict__ Wq, const float* __restrict__ Wk,
                         const float* __restrict__ Wv, const float* __restrict__ Ws,
                         unsigned* __restrict__ wp) {
    int tid  = blockIdx.x * 256 + threadIdx.x;   // 4*4*8*256 = 32768 exact
    int j    = tid & 7;
    int lane = (tid >> 3) & 31;
    int nt   = (tid >> 8) & 7;
    int kt   = (tid >> 11) & 3;
    int mat  = tid >> 13;
    const float* W = (mat == 0) ? Wq : (mat == 1) ? Wk : (mat == 2) ? Wv : Ws;
    int n  = nt * 16 + (lane & 15);
    int k0 = kt * 32 + ((j >> 2) << 4) + ((lane >> 4) << 3) + ((j & 3) << 1);
    float w0 = W[k0 * DDIM + n];
    float w1 = W[(k0 + 1) * DDIM + n];
    wp[tid] = (unsigned)f2bf(w0) | ((unsigned)f2bf(w1) << 16);
}

// ---- fused Q/K/V/Skip projection GEMM via v_wmma_f32_16x16x32_bf16 ---------
// grid = (3125 mtiles, 4 matrices); 8 waves/block, one 16x16 N-tile per wave.
__global__ void k_gemm(const unsigned* __restrict__ ap, const unsigned* __restrict__ wp,
                       const float* __restrict__ bq, const float* __restrict__ bk,
                       const float* __restrict__ bv, const float* __restrict__ bs,
                       float* __restrict__ q, float* __restrict__ k,
                       float* __restrict__ v, float* __restrict__ s) {
    int mt   = blockIdx.x;
    int mat  = blockIdx.y;
    int wave = threadIdx.x >> 5;                 // n-tile index 0..7
    int lane = threadIdx.x & 31;
    const float* bias = (mat == 0) ? bq : (mat == 1) ? bk : (mat == 2) ? bv : bs;
    float*       out  = (mat == 0) ? q  : (mat == 1) ? k  : (mat == 2) ? v  : s;

    union { v16bf bv16; u32x4 u4[2]; } A, B;
    v8f acc = {0.f, 0.f, 0.f, 0.f, 0.f, 0.f, 0.f, 0.f};

#pragma unroll
    for (int kt = 0; kt < KTILES; ++kt) {
        const u32x4* aptr = (const u32x4*)(ap + (mt * 4 + kt) * 256) + (lane << 1);
        A.u4[0] = aptr[0];
        A.u4[1] = aptr[1];
        const u32x4* bptr = (const u32x4*)(wp + ((mat * 4 + kt) * 8 + wave) * 256) + (lane << 1);
        B.u4[0] = bptr[0];
        B.u4[1] = bptr[1];
        acc = __builtin_amdgcn_wmma_f32_16x16x32_bf16(
            false, A.bv16, false, B.bv16, (short)0, acc, false, false);
    }

    int col   = wave * 16 + (lane & 15);
    int rbase = mt * 16 + ((lane >> 4) << 3);
    float bval = bias[col];
#pragma unroll
    for (int r = 0; r < 8; ++r)
        out[(rbase + r) * DDIM + col] = acc[r] + bval;
}

// ---- per-layer segment-softmax state init ----------------------------------
__global__ void k_init_layer(unsigned* __restrict__ amax, float* __restrict__ den) {
    int tid = blockIdx.x * 256 + threadIdx.x;
    if (tid < NN * HH) { amax[tid] = 0u; den[tid] = 0.f; }
}

// ---- attention logits + segment max (one thread per (edge, head)) ----------
__global__ void k_alpha(const int* __restrict__ ei, const float* __restrict__ q,
                        const float* __restrict__ k, float* __restrict__ alpha,
                        unsigned* __restrict__ amax) {
    int tid = blockIdx.x * 256 + threadIdx.x;    // EE*HH exact
    int e = tid >> 3, h = tid & 7;
    int s = ei[e], d = ei[EE + e];
    const f32x4* qp = (const f32x4*)(q + (size_t)d * DDIM + h * CC);
    const f32x4* kp = (const f32x4*)(k + (size_t)s * DDIM + h * CC);
    float acc = 0.f;
#pragma unroll
    for (int i = 0; i < 4; ++i) {
        f32x4 a = qp[i], b = kp[i];
        acc += a.x * b.x + a.y * b.y + a.z * b.z + a.w * b.w;
    }
    acc *= 0.25f;                                // 1/sqrt(C), C=16
    alpha[tid] = acc;
    atomicMax(&amax[d * HH + h], fenc(acc));
}

// ---- exp(alpha - max) + segment sum ----------------------------------------
__global__ void k_expsum(const int* __restrict__ ei, float* __restrict__ alpha,
                         const unsigned* __restrict__ amax, float* __restrict__ den) {
    int tid = blockIdx.x * 256 + threadIdx.x;    // EE*HH exact
    int e = tid >> 3, h = tid & 7;
    int d = ei[EE + e];
    float ea = __expf(alpha[tid] - fdec(amax[d * HH + h]));
    alpha[tid] = ea;
    atomicAdd(&den[d * HH + h], ea);
}

// ---- weighted message scatter (one thread per (edge, feature)) -------------
__global__ void k_scatter(const int* __restrict__ ei, const float* __restrict__ alpha,
                          const float* __restrict__ den, const float* __restrict__ v,
                          float* __restrict__ agg) {
    int tid = blockIdx.x * 256 + threadIdx.x;    // EE*DDIM = 102,400,000 exact
    int e = tid >> 7, c = tid & 127, h = c >> 4;
    int s = ei[e], d = ei[EE + e];
    float attn = alpha[e * HH + h] / (den[d * HH + h] + 1e-16f);
    atomicAdd(&agg[(size_t)d * DDIM + c], v[(size_t)s * DDIM + c] * attn);
}

// ---- readout ---------------------------------------------------------------
__global__ void k_init_pool(float* __restrict__ out, unsigned* __restrict__ pmax) {
    int tid = blockIdx.x * 256 + threadIdx.x;    // 16384
    if (tid < GG * 2 * DDIM) out[tid] = 0.f;
    if (tid < GG * DDIM) pmax[tid] = 0x00800000u;   // enc(-FLT_MAX)
}

__global__ void k_pool(const float* __restrict__ x, const int* __restrict__ batch,
                       const float* __restrict__ wg, const float* __restrict__ bg,
                       float* __restrict__ out, unsigned* __restrict__ pmax) {
    int gid  = blockIdx.x * 256 + threadIdx.x;   // NN*32 = 1,600,000 exact
    int node = gid >> 5, lane = gid & 31;
    int g = batch[node];
    f32x4 xv = ((const f32x4*)(x + (size_t)node * DDIM))[lane];
    f32x4 wv = ((const f32x4*)wg)[lane];
    float p = xv.x * wv.x + xv.y * wv.y + xv.z * wv.z + xv.w * wv.w;
#pragma unroll
    for (int off = 16; off; off >>= 1) p += __shfl_xor(p, off, 32);
    float score = 1.f / (1.f + __expf(-(p + bg[0])));
    float*    po = out + g * (2 * DDIM);
    unsigned* pm = pmax + g * DDIM;
    int c0 = lane * 4;
    atomicAdd(po + c0 + 0, score * xv.x);  atomicMax(pm + c0 + 0, fenc(xv.x));
    atomicAdd(po + c0 + 1, score * xv.y);  atomicMax(pm + c0 + 1, fenc(xv.y));
    atomicAdd(po + c0 + 2, score * xv.z);  atomicMax(pm + c0 + 2, fenc(xv.z));
    atomicAdd(po + c0 + 3, score * xv.w);  atomicMax(pm + c0 + 3, fenc(xv.w));
}

__global__ void k_fin(const unsigned* __restrict__ pmax, float* __restrict__ out) {
    int tid = blockIdx.x * 256 + threadIdx.x;    // 8192
    if (tid < GG * DDIM) {
        int g = tid >> 7, c = tid & 127;
        out[g * (2 * DDIM) + DDIM + c] = fdec(pmax[tid]);
    }
}

// ---- host ------------------------------------------------------------------
extern "C" void kernel_launch(void* const* d_in, const int* in_sizes, int n_in,
                              void* d_out, int out_size, void* d_ws, size_t ws_size,
                              hipStream_t stream) {
    const float* x   = (const float*)d_in[0];
    const int*   ei  = (const int*)d_in[1];
    const int*   bat = (const int*)d_in[2];
    const float* Wq  = (const float*)d_in[3];
    const float* bq  = (const float*)d_in[4];
    const float* Wk  = (const float*)d_in[5];
    const float* bk  = (const float*)d_in[6];
    const float* Wv  = (const float*)d_in[7];
    const float* bv  = (const float*)d_in[8];
    const float* Ws  = (const float*)d_in[9];
    const float* bs  = (const float*)d_in[10];
    const float* wg  = (const float*)d_in[11];
    const float* bg  = (const float*)d_in[12];
    float* out = (float*)d_out;

    char* w = (char*)d_ws;
    float*    xA    = (float*)w;     w += (size_t)NN * DDIM * 4;
    float*    xB    = (float*)w;     w += (size_t)NN * DDIM * 4;
    float*    qb    = (float*)w;     w += (size_t)NN * DDIM * 4;
    float*    kb    = (float*)w;     w += (size_t)NN * DDIM * 4;
    float*    vb    = (float*)w;     w += (size_t)NN * DDIM * 4;
    unsigned* ap    = (unsigned*)w;  w += (size_t)NN * DDIM / 2 * 4;   // packed bf16 A
    unsigned* wp    = (unsigned*)w;  w += (size_t)4 * KTILES * NTILES * 256 * 4;
    float*    alpha = (float*)w;     w += (size_t)EE * HH * 4;
    unsigned* amax  = (unsigned*)w;  w += (size_t)NN * HH * 4;
    float*    den   = (float*)w;     w += (size_t)NN * HH * 4;
    unsigned* pmax  = (unsigned*)w;  w += (size_t)GG * DDIM * 4;

    const float* cur = x;        // layer input (input tensor for layer 0)
    float* nxt   = xB;           // layer output buffer
    float* other = xA;

    for (int l = 0; l < LL; ++l) {
        size_t wo = (size_t)l * DDIM * DDIM;
        size_t bo = (size_t)l * DDIM;
        k_pack_w<<<128, 256, 0, stream>>>(Wq + wo, Wk + wo, Wv + wo, Ws + wo, wp);
        k_pack_a<<<NN * DDIM / 2 / 256, 256, 0, stream>>>(cur, ap);
        k_gemm<<<dim3(MT, 4), 256, 0, stream>>>(ap, wp, bq + bo, bk + bo, bv + bo, bs + bo,
                                                qb, kb, vb, nxt);
        k_init_layer<<<(NN * HH + 255) / 256, 256, 0, stream>>>(amax, den);
        k_alpha<<<EE * HH / 256, 256, 0, stream>>>(ei, qb, kb, alpha, amax);
        k_expsum<<<EE * HH / 256, 256, 0, stream>>>(ei, alpha, amax, den);
        k_scatter<<<EE * DDIM / 256, 256, 0, stream>>>(ei, alpha, den, vb, nxt);
        cur = nxt;
        float* t = nxt; nxt = other; other = t;
    }

    k_init_pool<<<(GG * 2 * DDIM) / 256, 256, 0, stream>>>(out, pmax);
    k_pool<<<NN * 32 / 256, 256, 0, stream>>>(cur, bat, wg, bg, out, pmax);
    k_fin<<<(GG * DDIM) / 256, 256, 0, stream>>>(pmax, out);
}
